// CrossFusionModule_53206054863375
// MI455X (gfx1250) — matrix operand order repaired and checked by reading, not compiled, not run on hardware
//
#include <hip/hip_runtime.h>

typedef __attribute__((ext_vector_type(16))) __bf16 v16bf;
typedef __attribute__((ext_vector_type(8)))  float  v8f;

#define BB  512
#define SS  64
#define EE  768
#define DD  256
#define FIN 1024

// A-fragment (16-bit, 16x32 MxK): VGPR v holds K = kbase, kbase+1 for this lane half
__device__ __forceinline__ int a_kbase(int v, int hi) {
    return ((v < 4) ? (2 * v) : (2 * v + 8)) + 8 * hi;
}
// B-fragment (16-bit, 32x16 KxN): VGPR v holds rows K = kbase, kbase+1 (same K order as A) at N = lane%16
__device__ __forceinline__ int b_kbase(int v, int hi) {
    return ((v < 4) ? (2 * v) : (2 * v + 8)) + 8 * hi;
}

// gfx1250 async global->LDS copy (ASYNCcnt-tracked), 16 bytes per lane
__device__ __forceinline__ void async_copy16(const void* gsrc, uint32_t lds_byte) {
    asm volatile("global_load_async_to_lds_b128 %0, %1, off"
                 :: "v"(lds_byte), "v"(gsrc) : "memory");
}
__device__ __forceinline__ void wait_async0() {
    asm volatile("s_wait_asynccnt 0x0" ::: "memory");
}

// ---------------- Stage 0: one-time fp32 -> bf16 weight conversion (4 elems/thread)
__global__ void cvt_kernel(const float* __restrict__ src, __bf16* __restrict__ dst) {
    int i = (blockIdx.x * 256 + threadIdx.x) * 4;
#pragma unroll
    for (int k = 0; k < 4; ++k) dst[i + k] = (__bf16)src[i + k];
}

// ---------------- Stage 1: proj[m] = latent[m] @ Wp[m]^T + bp[m]
// wave: 16 rows x 128 cols (8 N-tiles); block: 8 waves = 64 rows, full D covered by 2 waves
__global__ void proj_kernel(const float* __restrict__ latent,  // [B*S, E] fp32
                            const __bf16* __restrict__ Wpb,    // [D, E] bf16
                            const float* __restrict__ bp,      // [D]
                            __bf16* __restrict__ proj)         // [B*S, D] bf16
{
    const int lane = threadIdx.x & 31;
    const int wave = threadIdx.x >> 5;
    const int hi   = lane >> 4;
    const int l15  = lane & 15;
    const int rowbase = blockIdx.x * 64 + (wave >> 1) * 16;
    const int colbase = (wave & 1) * 128;

    v8f c[8];
#pragma unroll
    for (int t = 0; t < 8; ++t) {
        float bias = bp[colbase + t * 16 + l15];
#pragma unroll
        for (int r = 0; r < 8; ++r) c[t][r] = bias;
    }
    const float* arow = latent + (size_t)(rowbase + l15) * EE;
    for (int k0 = 0; k0 < EE; k0 += 32) {
        v16bf a;
#pragma unroll
        for (int v = 0; v < 8; ++v) {
            int kb = k0 + a_kbase(v, hi);
            a[2 * v]     = (__bf16)arow[kb];
            a[2 * v + 1] = (__bf16)arow[kb + 1];
        }
#pragma unroll
        for (int t = 0; t < 8; ++t) {
            const __bf16* brow = Wpb + (size_t)(colbase + t * 16 + l15) * EE;
            v16bf bfr;
#pragma unroll
            for (int v = 0; v < 8; ++v) {
                int kb = k0 + b_kbase(v, hi);
                bfr[2 * v]     = brow[kb];
                bfr[2 * v + 1] = brow[kb + 1];
            }
            c[t] = __builtin_amdgcn_wmma_f32_16x16x32_bf16(false, a, false, bfr,
                                                           (short)0, c[t], false, false);
        }
    }
#pragma unroll
    for (int t = 0; t < 8; ++t)
#pragma unroll
        for (int r = 0; r < 8; ++r) {
            int row = rowbase + r + 8 * hi;
            int col = colbase + t * 16 + l15;
            proj[(size_t)row * DD + col] = (__bf16)c[t][r];
        }
}

// ---------------- Stage 2: acorr = anchor @ corr  (all bf16)
__global__ void acorr_kernel(const __bf16* __restrict__ anchor, // [B*S, D]
                             const __bf16* __restrict__ corrb,  // [D, D] bf16
                             __bf16* __restrict__ acorr)        // [B*S, D]
{
    const int lane = threadIdx.x & 31;
    const int wave = threadIdx.x >> 5;
    const int hi   = lane >> 4;
    const int l15  = lane & 15;
    const int rowbase = blockIdx.x * 64 + (wave >> 1) * 16;
    const int colbase = (wave & 1) * 128;

    v8f c[8] = {};
    const __bf16* arow = anchor + (size_t)(rowbase + l15) * DD;
    for (int k0 = 0; k0 < DD; k0 += 32) {
        v16bf a;
#pragma unroll
        for (int v = 0; v < 8; ++v) {
            int kb = k0 + a_kbase(v, hi);
            a[2 * v]     = arow[kb];
            a[2 * v + 1] = arow[kb + 1];
        }
#pragma unroll
        for (int t = 0; t < 8; ++t) {
            int n = colbase + t * 16 + l15;
            v16bf bfr;
#pragma unroll
            for (int v = 0; v < 8; ++v) {
                int kb = k0 + b_kbase(v, hi);
                bfr[2 * v]     = corrb[(size_t)kb * DD + n];
                bfr[2 * v + 1] = corrb[(size_t)(kb + 1) * DD + n];
            }
            c[t] = __builtin_amdgcn_wmma_f32_16x16x32_bf16(false, a, false, bfr,
                                                           (short)0, c[t], false, false);
        }
    }
#pragma unroll
    for (int t = 0; t < 8; ++t)
#pragma unroll
        for (int r = 0; r < 8; ++r) {
            int row = rowbase + r + 8 * hi;
            int col = colbase + t * 16 + l15;
            acorr[(size_t)row * DD + col] = (__bf16)c[t][r];
        }
}

// ---------------- Stage 3: cc[b,i,j] = sum_d acorr[b,i,d] * other[b,j,d]
__global__ void cc_kernel(const __bf16* __restrict__ acorr, // [B,S,D]
                          const __bf16* __restrict__ other, // [B,S,D]
                          float* __restrict__ cc)           // [B,S,S]
{
    const int b    = blockIdx.x;
    const int lane = threadIdx.x & 31;
    const int wave = threadIdx.x >> 5;
    const int hi   = lane >> 4;
    const int l15  = lane & 15;
    const __bf16* A = acorr + (size_t)b * SS * DD;
    const __bf16* O = other + (size_t)b * SS * DD;
    float* C = cc + (size_t)b * SS * SS;

    for (int t = wave; t < 16; t += 8) {
        int tm = t >> 2, tn = t & 3;
        v8f c = {};
        const __bf16* arow = A + (size_t)(tm * 16 + l15) * DD;
        const __bf16* brow = O + (size_t)(tn * 16 + l15) * DD;
        for (int k0 = 0; k0 < DD; k0 += 32) {
            v16bf a, bfr;
#pragma unroll
            for (int v = 0; v < 8; ++v) {
                int ka = k0 + a_kbase(v, hi);
                a[2 * v]     = arow[ka];
                a[2 * v + 1] = arow[ka + 1];
                int kb = k0 + b_kbase(v, hi);
                bfr[2 * v]     = brow[kb];
                bfr[2 * v + 1] = brow[kb + 1];
            }
            c = __builtin_amdgcn_wmma_f32_16x16x32_bf16(false, a, false, bfr,
                                                        (short)0, c, false, false);
        }
#pragma unroll
        for (int r = 0; r < 8; ++r) {
            int i = tm * 16 + r + 8 * hi;
            C[(size_t)i * SS + tn * 16 + l15] = c[r];
        }
    }
}

// ---------------- Stage 4: async-stage proj tiles, softmaxes, attention GEMMs
// Computes T[j,d] = sum_i a_att[i,j]*anchor[i,d]  (+anchor[j,d])   -> fused[b,j, p*512+d]
//          T2[i,d] = sum_j o_att[i,j]*other[j,d]  (+other[i,d])    -> fused[b,i, p*512+256+d]
__global__ void attn_kernel(const __bf16* __restrict__ anchor, // proj0 [B,S,D]
                            const __bf16* __restrict__ other,  // proj_p [B,S,D]
                            const float* __restrict__ cc,      // [B,S,S]
                            __bf16* __restrict__ fused,        // [B,S,1024]
                            int pair)
{
    __shared__ __bf16 anS[SS * DD];    // 32 KB anchor tile
    __shared__ __bf16 otS[SS * DD];    // 32 KB other tile
    __shared__ float  ccs[SS * 65];    // padded: conflict-free row & column softmax
    __shared__ __bf16 aatT[SS * SS];   // a_att transposed: [j][i]
    __shared__ __bf16 oatS[SS * SS];   // o_att: [i][j]
    const int b   = blockIdx.x;
    const int tid = threadIdx.x;
    const __bf16* Ag = anchor + (size_t)b * SS * DD;
    const __bf16* Og = other + (size_t)b * SS * DD;
    const float*  Cg = cc + (size_t)b * SS * SS;

    // async global->LDS staging of both 64x256 bf16 tiles (16B chunks, 8 rounds each)
    uint32_t anB = (uint32_t)(uintptr_t)(&anS[0]);
    uint32_t otB = (uint32_t)(uintptr_t)(&otS[0]);
#pragma unroll
    for (int it = 0; it < 8; ++it) {
        int chunk = it * 256 + tid;                  // 2048 chunks of 16 B
        async_copy16(Ag + chunk * 8, anB + chunk * 16);
        async_copy16(Og + chunk * 8, otB + chunk * 16);
    }
    for (int idx = tid; idx < SS * SS; idx += 256)   // cc into padded LDS
        ccs[(idx >> 6) * 65 + (idx & 63)] = Cg[idx];
    wait_async0();
    __syncthreads();

    if (tid < 64) {                   // column softmax (axis=1) -> a_att^T[j, :]
        int j = tid;
        float m = -1e30f;
        for (int i = 0; i < 64; ++i) m = fmaxf(m, ccs[i * 65 + j]);
        float s = 0.f;
        for (int i = 0; i < 64; ++i) s += __expf(ccs[i * 65 + j] - m);
        float inv = 1.f / s;
        for (int i = 0; i < 64; ++i)
            aatT[j * 64 + i] = (__bf16)(__expf(ccs[i * 65 + j] - m) * inv);
    } else if (tid < 128) {           // row softmax (axis=2) -> o_att[i, :]
        int i = tid - 64;
        float m = -1e30f;
        for (int j = 0; j < 64; ++j) m = fmaxf(m, ccs[i * 65 + j]);
        float s = 0.f;
        for (int j = 0; j < 64; ++j) s += __expf(ccs[i * 65 + j] - m);
        float inv = 1.f / s;
        for (int j = 0; j < 64; ++j)
            oatS[i * 64 + j] = (__bf16)(__expf(ccs[i * 65 + j] - m) * inv);
    }
    __syncthreads();

    const int lane = tid & 31, wave = tid >> 5;
    const int hi = lane >> 4, l15 = lane & 15;
    __bf16* F = fused + (size_t)b * SS * FIN + pair * 512;

    for (int tt = 0; tt < 16; ++tt) {   // 128 tiles total (64 anchor-path + 64 other-path)
        int t = wave * 16 + tt;
        bool isO = (t >= 64);
        int t2 = isO ? t - 64 : t;
        int dn = t2 >> 2;               // N tile over d (16 tiles)
        int jm = t2 & 3;                // M tile over j (anchor) / i (other)
        const __bf16* Bsrc  = isO ? otS : anS;   // proj tile in LDS (B operand)
        const __bf16* Afrag = isO ? oatS : aatT; // attention matrix in LDS (A operand)
        int mrow = jm * 16 + l15;
        int ncol = dn * 16 + l15;
        v8f c = {};
        for (int k0 = 0; k0 < SS; k0 += 32) {
            v16bf a, bfr;
#pragma unroll
            for (int v = 0; v < 8; ++v) {
                int ka = k0 + a_kbase(v, hi);
                a[2 * v]     = Afrag[mrow * 64 + ka];     // consecutive in LDS
                a[2 * v + 1] = Afrag[mrow * 64 + ka + 1];
                int kb = k0 + b_kbase(v, hi);
                bfr[2 * v]     = Bsrc[kb * DD + ncol];
                bfr[2 * v + 1] = Bsrc[(kb + 1) * DD + ncol];
            }
            c = __builtin_amdgcn_wmma_f32_16x16x32_bf16(false, a, false, bfr,
                                                        (short)0, c, false, false);
        }
#pragma unroll
        for (int r = 0; r < 8; ++r) {
            int row = jm * 16 + r + 8 * hi;                       // j or i
            float val = c[r] + (float)Bsrc[row * DD + ncol];      // + transpose term
            F[(size_t)row * FIN + (isO ? 256 : 0) + dn * 16 + l15] = (__bf16)val;
        }
    }
}

// ---------------- Stage 5: h = fused @ Wb^T + bb, layernorm(64), gamma/beta, relu
__global__ void head_kernel(const __bf16* __restrict__ fused, // [B*S, 1024]
                            const __bf16* __restrict__ Wbb,   // [64, 1024] bf16
                            const float* __restrict__ bb,
                            const float* __restrict__ gamma,
                            const float* __restrict__ beta,
                            float* __restrict__ out)          // [B*S, 64]
{
    const int lane = threadIdx.x & 31;
    const int wave = threadIdx.x >> 5;
    const int hi   = lane >> 4;
    const int l15  = lane & 15;
    const int rowbase = blockIdx.x * 128 + wave * 16;

    v8f c[4];
    float g[4], be[4];
#pragma unroll
    for (int nt = 0; nt < 4; ++nt) {
        int n = nt * 16 + l15;
        float bias = bb[n];
        g[nt] = gamma[n]; be[nt] = beta[n];
#pragma unroll
        for (int r = 0; r < 8; ++r) c[nt][r] = bias;
    }
    const __bf16* arow = fused + (size_t)(rowbase + l15) * FIN;
    for (int k0 = 0; k0 < FIN; k0 += 32) {
        v16bf a;
#pragma unroll
        for (int v = 0; v < 8; ++v) {
            int kb = k0 + a_kbase(v, hi);
            a[2 * v]     = arow[kb];
            a[2 * v + 1] = arow[kb + 1];
        }
#pragma unroll
        for (int nt = 0; nt < 4; ++nt) {
            const __bf16* brow = Wbb + (size_t)(nt * 16 + l15) * FIN;
            v16bf bfr;
#pragma unroll
            for (int v = 0; v < 8; ++v) {
                int kb = k0 + b_kbase(v, hi);
                bfr[2 * v]     = brow[kb];
                bfr[2 * v + 1] = brow[kb + 1];
            }
            c[nt] = __builtin_amdgcn_wmma_f32_16x16x32_bf16(false, a, false, bfr,
                                                            (short)0, c[nt], false, false);
        }
    }
    // per-row layernorm: row (r,hi) values live in c[0..3][r] across the 16-lane group
#pragma unroll
    for (int r = 0; r < 8; ++r) {
        float s  = c[0][r] + c[1][r] + c[2][r] + c[3][r];
        float sq = c[0][r]*c[0][r] + c[1][r]*c[1][r] + c[2][r]*c[2][r] + c[3][r]*c[3][r];
#pragma unroll
        for (int msk = 1; msk <= 8; msk <<= 1) {
            s  += __shfl_xor(s,  msk, 32);
            sq += __shfl_xor(sq, msk, 32);
        }
        float mean = s * (1.f / 64.f);
        float var  = sq * (1.f / 64.f) - mean * mean;
        float rstd = rsqrtf(var + 1e-5f);
        int row = rowbase + r + 8 * hi;
#pragma unroll
        for (int nt = 0; nt < 4; ++nt) {
            float v = (c[nt][r] - mean) * rstd * g[nt] + be[nt];
            out[(size_t)row * 64 + nt * 16 + l15] = fmaxf(v, 0.f);
        }
    }
}

extern "C" void kernel_launch(void* const* d_in, const int* in_sizes, int n_in,
                              void* d_out, int out_size, void* d_ws, size_t ws_size,
                              hipStream_t stream) {
    (void)in_sizes; (void)n_in; (void)out_size; (void)ws_size;
    // setup_inputs() dict order: latent0,Wp0,bp0, latent1,Wp1,bp1, latent2,Wp2,bp2,
    //                            corr, Wb, bb, gamma, beta
    const float* latent[3] = {(const float*)d_in[0], (const float*)d_in[3], (const float*)d_in[6]};
    const float* Wp[3]     = {(const float*)d_in[1], (const float*)d_in[4], (const float*)d_in[7]};
    const float* bp[3]     = {(const float*)d_in[2], (const float*)d_in[5], (const float*)d_in[8]};
    const float* corr  = (const float*)d_in[9];
    const float* Wb    = (const float*)d_in[10];
    const float* bb    = (const float*)d_in[11];
    const float* gamma = (const float*)d_in[12];
    const float* beta  = (const float*)d_in[13];
    float* out = (float*)d_out;

    const size_t PROJ = (size_t)BB * SS * DD;            // 8,388,608 elems
    __bf16* proj0 = (__bf16*)d_ws;
    __bf16* proj1 = proj0 + PROJ;
    __bf16* proj2 = proj1 + PROJ;
    __bf16* acorr = proj2 + PROJ;
    __bf16* fused = acorr + PROJ;                        // [B,S,1024] bf16
    float*  cc0   = (float*)(fused + (size_t)BB * SS * FIN);
    float*  cc1   = cc0 + (size_t)BB * SS * SS;
    __bf16* wpb   = (__bf16*)(cc1 + (size_t)BB * SS * SS);  // 3x[D,E] bf16
    __bf16* corrb = wpb + (size_t)3 * DD * EE;
    __bf16* wbb   = corrb + (size_t)DD * DD;             // total ws use ~152.5 MB

    dim3 blk(256);
    __bf16* projs[3] = {proj0, proj1, proj2};
    for (int m = 0; m < 3; ++m)                           // weights -> bf16 (once)
        cvt_kernel<<<dim3((DD * EE) / 1024), blk, 0, stream>>>(Wp[m], wpb + (size_t)m * DD * EE);
    cvt_kernel<<<dim3((DD * DD) / 1024), blk, 0, stream>>>(corr, corrb);
    cvt_kernel<<<dim3((64 * FIN) / 1024), blk, 0, stream>>>(Wb, wbb);

    for (int m = 0; m < 3; ++m)
        proj_kernel<<<dim3((BB * SS) / 64), blk, 0, stream>>>(latent[m], wpb + (size_t)m * DD * EE,
                                                              bp[m], projs[m]);
    acorr_kernel<<<dim3((BB * SS) / 64), blk, 0, stream>>>(proj0, corrb, acorr);
    cc_kernel<<<dim3(BB), blk, 0, stream>>>(acorr, proj1, cc0);
    cc_kernel<<<dim3(BB), blk, 0, stream>>>(acorr, proj2, cc1);
    attn_kernel<<<dim3(BB), blk, 0, stream>>>(proj0, proj1, cc0, fused, 0);
    attn_kernel<<<dim3(BB), blk, 0, stream>>>(proj0, proj2, cc1, fused, 1);
    head_kernel<<<dim3((BB * SS) / 128), blk, 0, stream>>>(fused, Wb ? wbb : wbb, bb, gamma, beta, out);
}
